// GroupedQueryAttention_71425306132778
// MI455X (gfx1250) — matrix operand order repaired and checked by reading, not compiled, not run on hardware
//
#include <hip/hip_runtime.h>
#include <cstdint>

#define B_  2
#define T_  2048
#define C_  2048
#define H_  16
#define G_  4
#define D_  128
#define KV_ 512
#define HPG (H_ / G_)

typedef __attribute__((ext_vector_type(16))) __bf16        v16bf;
typedef __attribute__((ext_vector_type(8)))  float         v8f;
typedef __attribute__((ext_vector_type(4)))  unsigned int  su4;
typedef __attribute__((ext_vector_type(8)))  unsigned int  su8;

union BF16x16 { v16bf v; unsigned short s[16]; uint4 q[2]; };

__device__ __forceinline__ unsigned short f2bf(float f) {
  union { float f; unsigned u; } x; x.f = f;
  unsigned r = x.u + 0x7FFFu + ((x.u >> 16) & 1u);   // round-to-nearest-even
  return (unsigned short)(r >> 16);
}

__device__ __forceinline__ v8f wmma_bf16(v16bf a, v16bf b, v8f c) {
  // D = A(16x32 bf16) * B(32x16 bf16) + C(16x16 f32)
  return __builtin_amdgcn_wmma_f32_16x16x32_bf16(false, a, false, b, (short)0, c, false, false);
}

// ---------------------------------------------------------------- convert
__global__ void cvt_f32_to_bf16(const float* __restrict__ in,
                                unsigned short* __restrict__ out, int n) {
  for (int i = blockIdx.x * blockDim.x + threadIdx.x; i < n;
       i += gridDim.x * blockDim.x)
    out[i] = f2bf(in[i]);
}

// ---------------------------------------------------------------- GEMM
// out[M,N] = A[M,K](bf16) @ Wt[N,K]^T (bf16) + bias, fp32 out.
// Register-blocked: one wave computes a 32x64 tile (2 M-frags x 4 N-frags),
// 8 WMMAs per K-step for 6 fragment loads -> ~3x operand reuse vs 16x16.
#define MT_ 2
#define NT_ 4
__global__ void gemm_bf16_blk(const unsigned short* __restrict__ A,
                              const unsigned short* __restrict__ Wt,
                              const float* __restrict__ bias,
                              float* __restrict__ out, int M, int N, int K) {
  const int lane = threadIdx.x & 31;
  const int half = lane >> 4;      // lane group (0: lanes 0-15, 1: 16-31)
  const int l16  = lane & 15;
  const int n0 = blockIdx.x * (16 * NT_);
  const int m0 = blockIdx.y * (16 * MT_);

  const unsigned short* abase = A  + (size_t)(m0 + l16) * K + half * 8;
  const unsigned short* bbase = Wt + (size_t)(n0 + l16) * K + half * 8;

  v8f acc[MT_][NT_] = {};
  for (int kb = 0; kb < K; kb += 32) {
    v16bf af[MT_], bf[NT_];
#pragma unroll
    for (int mt = 0; mt < MT_; ++mt) {
      const unsigned short* p = abase + (size_t)mt * 16 * K + kb;
      __builtin_prefetch(p + 128, 0, 1);
      BF16x16 u;
      u.q[0] = *(const uint4*)p;          // K = kb + half*8 + 0..7
      u.q[1] = *(const uint4*)(p + 16);   // K = kb + 16 + half*8 + 0..7
      af[mt] = u.v;
    }
#pragma unroll
    for (int nt = 0; nt < NT_; ++nt) {
      const unsigned short* p = bbase + (size_t)nt * 16 * K + kb;
      __builtin_prefetch(p + 128, 0, 1);
      BF16x16 u;
      u.q[0] = *(const uint4*)p;          // K = kb + half*8 + 0..7
      u.q[1] = *(const uint4*)(p + 16);   // K = kb + 16 + half*8 + 0..7
      bf[nt] = u.v;
    }
#pragma unroll
    for (int mt = 0; mt < MT_; ++mt)
#pragma unroll
      for (int nt = 0; nt < NT_; ++nt)
        acc[mt][nt] = wmma_bf16(af[mt], bf[nt], acc[mt][nt]);
  }

#pragma unroll
  for (int mt = 0; mt < MT_; ++mt) {
#pragma unroll
    for (int nt = 0; nt < NT_; ++nt) {
      const float bv = bias ? bias[n0 + nt * 16 + l16] : 0.0f;
      float* orow = out + (size_t)(m0 + mt * 16 + half * 8) * N
                        + n0 + nt * 16 + l16;
#pragma unroll
      for (int r = 0; r < 8; ++r)
        orow[(size_t)r * N] = acc[mt][nt][r] + bv;
    }
  }
}

// ------------------------------------------------- RMSNorm + RoPE + transpose
// in:  [B, T, nh, D] fp32   out: [B, nh, T, D] bf16
__global__ void norm_rope_tr(const float* __restrict__ in,
                             const float* __restrict__ w,
                             unsigned short* __restrict__ out,
                             int nh, int doRope) {
  __shared__ float sx[D_];
  __shared__ float red[4];
  const int idx = blockIdx.x;
  const int h = idx % nh;
  const int t = (idx / nh) % T_;
  const int b = idx / (nh * T_);
  const int d = threadIdx.x;

  float x = in[(((size_t)b * T_ + t) * nh + h) * D_ + d];
  float o;
  if (doRope) {
    float ss = x * x;                                  // ||x||^2 over D=128
#pragma unroll
    for (int off = 16; off; off >>= 1) ss += __shfl_xor(ss, off);
    if ((threadIdx.x & 31) == 0) red[threadIdx.x >> 5] = ss;
    __syncthreads();
    ss = red[0] + red[1] + red[2] + red[3];
    float rms = sqrtf(ss / (float)D_);
    sx[d] = w[d] * x / (rms + 1e-6f);
    __syncthreads();
    const int j = d & 63;
    // inv_freq = 10000^(-2j/128)
    float fr = (float)t * __expf(-9.210340371976184f * (float)(2 * j) / (float)D_);
    float c, s;
    __sincosf(fr, &s, &c);
    o = (d < 64) ? (sx[j] * c + sx[j + 64] * s)
                 : (-sx[j] * s + sx[j + 64] * c);
  } else {
    o = x;
  }
  out[(((size_t)b * nh + h) * T_ + t) * D_ + d] = f2bf(o);
}

// ---------------------------------------------------------------- attention
// Q: [B,H,T,D] bf16   K,V: [B,G,T,D] bf16   Y: [B,T,H,D] fp32
// One wave per (b, h, 16-row query tile). Flash-style online softmax.
// V tiles are staged into LDS by the Tensor Data Mover (TDM), overlapped
// with the softmax VALU work, and synchronized with s_wait_tensorcnt.
__global__ void attn_kernel(const unsigned short* __restrict__ Q,
                            const unsigned short* __restrict__ Kc,
                            const unsigned short* __restrict__ Vc,
                            float* __restrict__ Y) {
  __shared__ unsigned short Vs[32 * D_];   // V tile (32 keys x 128)
  __shared__ float          Ps[16 * 32];   // P tile (16 rows x 32 keys)

  const int lane = threadIdx.x & 31;
  const int half = lane >> 4;
  const int l16  = lane & 15;
  const int q0 = blockIdx.x * 16;
  const int h  = blockIdx.y;
  const int b  = blockIdx.z;
  const int g  = h / HPG;

  const unsigned short* Qh = Q  + ((size_t)(b * H_ + h) * T_) * D_;
  const unsigned short* Kh = Kc + ((size_t)(b * G_ + g) * T_) * D_;
  const unsigned short* Vh = Vc + ((size_t)(b * G_ + g) * T_) * D_;

  // LDS byte offset of Vs (flat shared address low 32 bits == LDS_ADDR)
  const unsigned vs_lds = (unsigned)(size_t)&Vs[0];

  // Q fragments, resident for the whole row of key tiles (D=128 -> 4 K-steps)
  v16bf qf[4];
#pragma unroll
  for (int kk = 0; kk < 4; ++kk) {
    const unsigned short* p = Qh + (size_t)(q0 + l16) * D_ + kk * 32 + half * 8;
    BF16x16 u; u.q[0] = *(const uint4*)p; u.q[1] = *(const uint4*)(p + 16);
    qf[kk] = u.v;
  }

  v8f o[8] = {};                     // output accumulators (16 x 128)
  float mrow[8], lrow[8];
#pragma unroll
  for (int r = 0; r < 8; ++r) { mrow[r] = -3.0e38f; lrow[r] = 0.0f; }

  const float iscale = 0.08838834764831845f;   // 1/sqrt(D)
  const int kend = q0 + 16;                    // causal bound

  for (int kb0 = 0; kb0 < kend; kb0 += 32) {
    // ---- kick off TDM: DMA the 32x128 bf16 V tile into LDS.
    // D# group0: count=1 | lds_addr | global_addr[56:0] | type=2
    // D# group1: data_size=2B, tensor_dim0=128, tensor_dim1=32,
    //            tile_dim0=128, tile_dim1=32, tensor_dim0_stride=128
    {
      const unsigned long long ga =
          (unsigned long long)(size_t)(Vh + (size_t)kb0 * D_);
      su4 g0 = { 1u, vs_lds, (unsigned)ga,
                 (unsigned)((ga >> 32) & 0x01FFFFFFu) | 0x80000000u };
      su8 g1 = { 0x00010000u,          // wg_mask=0, data_size=1 (2 bytes)
                 128u << 16,           // tensor_dim0 = 128 (bits 79:48)
                 32u  << 16,           // tensor_dim1 = 32  (bits 111:80)
                 128u << 16,           // tile_dim0   = 128 (bits 127:112)
                 32u,                  // tile_dim1   = 32  (bits 143:128)
                 128u,                 // tensor_dim0_stride = 128 (bits 207:160)
                 0u, 0u };
      asm volatile("s_wait_dscnt 0x0" ::: "memory");  // WAR vs prior Vs reads
      asm volatile("tensor_load_to_lds %0, %1" :: "s"(g0), "s"(g1) : "memory");
    }

    // ---- S = Q * K^T for two 16-key subtiles (overlaps with TDM)
    v8f s[2];
#pragma unroll
    for (int nt = 0; nt < 2; ++nt) {
      v8f acc = {};
#pragma unroll
      for (int kk = 0; kk < 4; ++kk) {
        const unsigned short* p =
            Kh + (size_t)(kb0 + nt * 16 + l16) * D_ + kk * 32 + half * 8;
        BF16x16 u; u.q[0] = *(const uint4*)p; u.q[1] = *(const uint4*)(p + 16);
        acc = wmma_bf16(qf[kk], u.v, acc);
      }
      s[nt] = acc;
    }

    // ---- mask, soft-cap, online softmax; write P to LDS
#pragma unroll
    for (int r = 0; r < 8; ++r) {
      const int row = q0 + half * 8 + r;
      float s0 = s[0][r] * iscale;
      float s1 = s[1][r] * iscale;
      if (kb0 + l16      > row) s0 = -1.0e30f;   // causal mask (pre-cap)
      if (kb0 + 16 + l16 > row) s1 = -1.0e30f;
      s0 = 50.0f * tanhf(s0 * 0.02f);            // soft cap (masked -> -50)
      s1 = 50.0f * tanhf(s1 * 0.02f);
      float rm = fmaxf(s0, s1);
#pragma unroll
      for (int off = 8; off; off >>= 1) rm = fmaxf(rm, __shfl_xor(rm, off));
      const float mnew = fmaxf(mrow[r], rm);
      const float corr = __expf(mrow[r] - mnew);
      const float p0 = __expf(s0 - mnew);
      const float p1 = __expf(s1 - mnew);
      float ps = p0 + p1;
#pragma unroll
      for (int off = 8; off; off >>= 1) ps += __shfl_xor(ps, off);
      lrow[r] = lrow[r] * corr + ps;
      mrow[r] = mnew;
#pragma unroll
      for (int i = 0; i < 8; ++i) o[i][r] *= corr;
      Ps[(half * 8 + r) * 32 + l16]      = p0;
      Ps[(half * 8 + r) * 32 + 16 + l16] = p1;
    }
    __syncthreads();

    // ---- O += P * V : re-fragment P (C-layout -> A-layout) through LDS
    BF16x16 pf;
#pragma unroll
    for (int j = 0; j < 16; ++j) {
      const int kc = ((j < 8) ? j : (j + 8)) + half * 8;
      pf.s[j] = f2bf(Ps[l16 * 32 + kc]);
    }

    __builtin_amdgcn_s_wait_tensorcnt(0);        // V tile landed in LDS
#pragma unroll
    for (int n8 = 0; n8 < 8; ++n8) {
      BF16x16 bvf;
#pragma unroll
      for (int j = 0; j < 16; ++j)
        bvf.s[j] = Vs[(((j < 8) ? j : (j + 8)) + half * 8) * D_ + n8 * 16 + l16];
      o[n8] = wmma_bf16(pf.v, bvf.v, o[n8]);
    }
    __syncthreads();
  }

  // ---- normalize and write y in [B, T, H*D] layout for the O-projection
#pragma unroll
  for (int r = 0; r < 8; ++r) {
    const int row = q0 + half * 8 + r;
    const float inv = 1.0f / lrow[r];
    float* yrow = Y + (((size_t)b * T_ + row) * H_ + h) * D_ + l16;
#pragma unroll
    for (int n8 = 0; n8 < 8; ++n8)
      yrow[n8 * 16] = o[n8][r] * inv;
  }
}

// ---------------------------------------------------------------- launcher
extern "C" void kernel_launch(void* const* d_in, const int* in_sizes, int n_in,
                              void* d_out, int out_size, void* d_ws, size_t ws_size,
                              hipStream_t stream) {
  (void)in_sizes; (void)n_in; (void)out_size; (void)ws_size;
  const float* x  = (const float*)d_in[0];
  const float* Wq = (const float*)d_in[1];
  const float* Wk = (const float*)d_in[2];
  const float* Wv = (const float*)d_in[3];
  const float* Wo = (const float*)d_in[4];
  const float* bo = (const float*)d_in[5];
  const float* qw = (const float*)d_in[6];
  const float* kw = (const float*)d_in[7];

  char* p = (char*)d_ws;
  auto alloc = [&](size_t bytes) {
    char* r = p; p += (bytes + 255) & ~(size_t)255; return r;
  };
  const size_t BT = (size_t)B_ * T_;

  unsigned short* xb  = (unsigned short*)alloc(BT * C_ * 2);
  unsigned short* wqb = (unsigned short*)alloc((size_t)C_ * C_ * 2);
  unsigned short* wkb = (unsigned short*)alloc((size_t)KV_ * C_ * 2);
  unsigned short* wvb = (unsigned short*)alloc((size_t)KV_ * C_ * 2);
  unsigned short* wob = (unsigned short*)alloc((size_t)C_ * C_ * 2);
  float*          qf  = (float*)alloc(BT * C_ * 4);     // later reused as Y
  float*          kf  = (float*)alloc(BT * KV_ * 4);
  float*          vf  = (float*)alloc(BT * KV_ * 4);
  unsigned short* qb2 = (unsigned short*)alloc(BT * C_ * 2);
  unsigned short* kb2 = (unsigned short*)alloc(BT * KV_ * 2);
  unsigned short* vb2 = (unsigned short*)alloc(BT * KV_ * 2);
  unsigned short* yb  = (unsigned short*)alloc(BT * C_ * 2);

  // 1) fp32 -> bf16 conversions
  cvt_f32_to_bf16<<<4096, 256, 0, stream>>>(x,  xb,  (int)(BT * C_));
  cvt_f32_to_bf16<<<4096, 256, 0, stream>>>(Wq, wqb, C_ * C_);
  cvt_f32_to_bf16<<<2048, 256, 0, stream>>>(Wk, wkb, KV_ * C_);
  cvt_f32_to_bf16<<<2048, 256, 0, stream>>>(Wv, wvb, KV_ * C_);
  cvt_f32_to_bf16<<<4096, 256, 0, stream>>>(Wo, wob, C_ * C_);

  // 2) QKV projections (register-blocked WMMA GEMMs, 32x64 tile per wave)
  gemm_bf16_blk<<<dim3(C_ / 64,  BT / 32), 32, 0, stream>>>(xb, wqb, nullptr, qf, (int)BT, C_,  C_);
  gemm_bf16_blk<<<dim3(KV_ / 64, BT / 32), 32, 0, stream>>>(xb, wkb, nullptr, kf, (int)BT, KV_, C_);
  gemm_bf16_blk<<<dim3(KV_ / 64, BT / 32), 32, 0, stream>>>(xb, wvb, nullptr, vf, (int)BT, KV_, C_);

  // 3) RMSNorm + RoPE + head-major transpose
  norm_rope_tr<<<B_ * T_ * H_, D_, 0, stream>>>(qf, qw, qb2, H_, 1);
  norm_rope_tr<<<B_ * T_ * G_, D_, 0, stream>>>(kf, kw, kb2, G_, 1);
  norm_rope_tr<<<B_ * T_ * G_, D_, 0, stream>>>(vf, nullptr, vb2, G_, 0);

  // 4) flash attention (qf fp32 buffer is dead now -> reuse as Y)
  float* yf = qf;
  attn_kernel<<<dim3(T_ / 16, H_, B_), 32, 0, stream>>>(qb2, kb2, vb2, yf);

  // 5) output projection + bias -> d_out (fp32)
  cvt_f32_to_bf16<<<4096, 256, 0, stream>>>(yf, yb, (int)(BT * C_));
  gemm_bf16_blk<<<dim3(C_ / 64, BT / 32), 32, 0, stream>>>(yb, wob, bo, (float*)d_out, (int)BT, C_, C_);
}
